// CustomMultiHeadedAttention_51934744544005
// MI455X (gfx1250) — matrix operand order repaired and checked by reading, not compiled, not run on hardware
//
#include <hip/hip_runtime.h>
#include <hip/hip_bf16.h>

typedef __attribute__((ext_vector_type(16))) _Float16 v16h;
typedef __attribute__((ext_vector_type(8)))  _Float16 v8h;
typedef __attribute__((ext_vector_type(2)))  _Float16 v2h;
typedef __attribute__((ext_vector_type(8)))  float    v8f;
typedef __attribute__((ext_vector_type(4)))  unsigned int u32x4;
typedef __attribute__((ext_vector_type(8)))  int      i32x8;
typedef __attribute__((ext_vector_type(4)))  int      i32x4;

#define WMMA_F32_16x16x32_F16(A, B, C) \
    __builtin_amdgcn_wmma_f32_16x16x32_f16(false, (A), false, (B), (short)0, (C), false, false)

#if __has_builtin(__builtin_amdgcn_tensor_load_to_lds) && \
    __has_builtin(__builtin_amdgcn_s_wait_tensorcnt)
#define HAVE_TDM 1
#else
#define HAVE_TDM 0
#endif

#if HAVE_TDM
// ---------------------------------------------------------------------------
// Tensor Data Mover: 2-D f16 tile DMA, global -> LDS.
// Descriptor bitfields per CDNA5 ISA:
//  group0: [1:0]=count=1, [63:32]=lds_addr, [120:64]=global_addr, [127:126]=type=2
//  group1: [17:16]=data_size(1->2B), [79:48]=tensor_dim0, [111:80]=tensor_dim1,
//          [127:112]=tile_dim0, [143:128]=tile_dim1, [207:160]=tensor_dim0_stride
//  groups 2/3 (and the trailing group on the 6-arg builtin): zero (tensor <= 2D)
__device__ __forceinline__ void tdm_load_2d_f16(unsigned int lds_off, const void* gaddr,
                                                unsigned int tensor_w, unsigned int tensor_h,
                                                unsigned int tile_w, unsigned int tile_h,
                                                unsigned int stride_w) {
    unsigned long long ga = (unsigned long long)(size_t)gaddr;
    u32x4 g0;
    g0[0] = 1u;                                        // count=1 (valid user D#)
    g0[1] = lds_off;                                   // lds_addr (bytes)
    g0[2] = (unsigned int)ga;                          // global_addr[31:0]
    g0[3] = (unsigned int)(ga >> 32) | (2u << 30);     // global_addr[56:32] | type=2
    i32x8 g1;
    g1[0] = (int)(1u << 16);                           // data_size = 2 bytes
    g1[1] = (int)((tensor_w & 0xFFFFu) << 16);         // tensor_dim0 lo16
    g1[2] = (int)((tensor_w >> 16) | ((tensor_h & 0xFFFFu) << 16));
    g1[3] = (int)((tensor_h >> 16) | (tile_w << 16));  // tensor_dim1 hi | tile_dim0
    g1[4] = (int)tile_h;                               // tile_dim1 (tile_dim2=0)
    g1[5] = (int)stride_w;                             // tensor_dim0_stride lo32
    g1[6] = 0;
    g1[7] = 0;
    i32x4 g2 = {0, 0, 0, 0};
    i32x4 g3 = {0, 0, 0, 0};
    i32x8 g4 = {0, 0, 0, 0, 0, 0, 0, 0};
    __builtin_amdgcn_tensor_load_to_lds(g0, g1, g2, g3, g4, 0);
}
#endif

// ---------------------------------------------------------------------------
// Fragment loaders (layouts per CDNA5 ISA §7.12.2, wave32)
//
// A (16x32 f16, M x K): lanes 0-15 hold row M=lane with K = {0..7,16..23};
// lanes 16-31 hold row M=lane-16 with K = {8..15,24..31}.
__device__ __forceinline__ v16h load_frag_a(const _Float16* base, int stride,
                                            int row0, int kbase, int lane) {
    const _Float16* p = base + (size_t)(row0 + (lane & 15)) * stride
                             + kbase + ((lane & 16) ? 8 : 0);
    v8h lo = *(const v8h*)p;
    v8h hi = *(const v8h*)(p + 16);
    v16h r;
#pragma unroll
    for (int i = 0; i < 8; ++i) { r[i] = lo[i]; r[i + 8] = hi[i]; }
    return r;
}

// B (32x16 f16, K x N): lane (N=lane&15) holds 16 K values,
// lanes 0-15 -> K={0..7,16..23}, lanes 16-31 -> K={8..15,24..31}. Source array must be
// K-major per column.
__device__ __forceinline__ v16h load_frag_b(const _Float16* base, int stride,
                                            int col0, int kbase, int lane) {
    const _Float16* p = base + (size_t)(col0 + (lane & 15)) * stride
                             + kbase + ((lane & 16) ? 8 : 0);
    v8h lo = *(const v8h*)p;
    v8h hi = *(const v8h*)(p + 16);
    v16h r;
#pragma unroll
    for (int i = 0; i < 8; ++i) { r[i] = lo[i]; r[i + 8] = hi[i]; }
    return r;
}

// ---------------------------------------------------------------------------
// Kernel 1: QKV = x @ w_attn + b_attn, fused RoPE, f16 per-head output.
// Block tile 128(M) x 128(N), K-step 32. 8 waves: wave = 32 rows x 64 cols
// (one full head) -> 2 A-frags x 4 B-frags = 8 WMMA per staging round.
__global__ __launch_bounds__(256) void qkv_rope_kernel(
        const float* __restrict__ x, const float* __restrict__ w,
        const float* __restrict__ b,
        _Float16* __restrict__ Q, _Float16* __restrict__ K,
        _Float16* __restrict__ V) {
    __shared__ __align__(16) _Float16 lds_a[128 * 32];   // [row][k]
    __shared__ __align__(16) _Float16 lds_bt[128 * 32];  // [col][k] transposed

    const int tid  = threadIdx.x;
    const int lane = tid & 31;
    const int wv   = tid >> 5;
    const int m0   = blockIdx.y * 128;
    const int n0   = blockIdx.x * 128;
    const int mt   = (wv & 3) * 32;   // wave: 32 rows
    const int nb   = (wv >> 2) * 64;  // wave: 64 cols (head-aligned)
    const int ncol = lane & 15;

    v8f acc[2][4] = {};

    for (int kk = 0; kk < 1024; kk += 32) {
        {   // stage A: x[m0..+127][kk..+31] -> lds_a (f32->f16), 2x b128 stores
            int r = tid >> 1, c = (tid & 1) * 16;
            const float* gp = x + (size_t)(m0 + r) * 1024 + kk + c;
            __builtin_prefetch(gp + 32, 0, 3);
            v8h h0, h1;
#pragma unroll
            for (int i = 0; i < 8; ++i) {
                h0[i] = (_Float16)gp[i];
                h1[i] = (_Float16)gp[8 + i];
            }
            *(v8h*)(lds_a + r * 32 + c)     = h0;
            *(v8h*)(lds_a + r * 32 + c + 8) = h1;
        }
        {   // stage B transposed: pair of k-rows packed -> ds_store_b32
            int r  = (tid >> 4) * 2;     // 0..30
            int c0 = (tid & 15) * 8;     // 0..120
            const float* g0 = w + (size_t)(kk + r) * 3072 + n0 + c0;
            const float* g1 = g0 + 3072;
            __builtin_prefetch(g0 + 32 * 3072, 0, 3);
#pragma unroll
            for (int i = 0; i < 8; ++i) {
                v2h p;
                p[0] = (_Float16)g0[i];
                p[1] = (_Float16)g1[i];
                *(v2h*)(lds_bt + (c0 + i) * 32 + r) = p;
            }
        }
        __syncthreads();

        v16h a0 = load_frag_a(lds_a, 32, mt, 0, lane);
        v16h a1 = load_frag_a(lds_a, 32, mt + 16, 0, lane);
#pragma unroll
        for (int nt = 0; nt < 4; ++nt) {
            v16h bf = load_frag_b(lds_bt, 32, nb + nt * 16, 0, lane);
            acc[0][nt] = WMMA_F32_16x16x32_F16(a0, bf, acc[0][nt]);
            acc[1][nt] = WMMA_F32_16x16x32_F16(a1, bf, acc[1][nt]);
        }
        __syncthreads();
    }

    // Epilogue: bias, RoPE (partner d^32 lives in acc[rt][nt^2] of the same
    // thread because each wave owns a full head), f16 per-head store.
    const int row_off = (lane & 16) ? 8 : 0;
    const int col0    = n0 + nb;           // head-aligned
    const int region  = col0 >> 10;        // 0=Q 1=K 2=V
    const int head    = (col0 & 1023) >> 6;
    _Float16* dst = (region == 0) ? Q : (region == 1) ? K : V;
    dst += (size_t)head * 4096 * 64;

#pragma unroll
    for (int rt = 0; rt < 2; ++rt) {
        const int mrow = m0 + mt + rt * 16 + row_off;
        float vals[4][8];
#pragma unroll
        for (int nt = 0; nt < 4; ++nt) {
            float bc = b[col0 + nt * 16 + ncol];
#pragma unroll
            for (int i = 0; i < 8; ++i) vals[nt][i] = acc[rt][nt][i] + bc;
        }
        if (region < 2) {  // RoPE
            float res[4][8];
#pragma unroll
            for (int nt = 0; nt < 4; ++nt) {
                int d  = nt * 16 + ncol;
                int dm = d & 31;
                float inv = __powf(10000.0f, -(float)dm * (1.0f / 32.0f));
                float sgn = (d < 32) ? -1.0f : 1.0f;
#pragma unroll
                for (int i = 0; i < 8; ++i) {
                    float th = (float)(mrow + i) * inv;
                    res[nt][i] = vals[nt][i] * __cosf(th)
                               + sgn * vals[nt ^ 2][i] * __sinf(th);
                }
            }
#pragma unroll
            for (int nt = 0; nt < 4; ++nt)
#pragma unroll
                for (int i = 0; i < 8; ++i) vals[nt][i] = res[nt][i];
        }
#pragma unroll
        for (int nt = 0; nt < 4; ++nt)
#pragma unroll
            for (int i = 0; i < 8; ++i)
                dst[(size_t)(mrow + i) * 64 + nt * 16 + ncol] = (_Float16)vals[nt][i];
    }
}

// ---------------------------------------------------------------------------
// Kernel 2: causal flash attention per head. Block = 1 head x 128 query rows,
// 8 waves x 16 rows. Key tiles of 64, f32 online softmax. K tile staged by
// the Tensor Data Mover (wave 0), V^T staged manually (TDM can't transpose).
__global__ __launch_bounds__(256) void attn_kernel(
        const _Float16* __restrict__ Q, const _Float16* __restrict__ K,
        const _Float16* __restrict__ V, _Float16* __restrict__ Y) {
    __shared__ __align__(16) _Float16 lds_k[64 * 64];      // [key][d]
    __shared__ __align__(16) _Float16 lds_vt[64 * 64];     // [d][key]
    __shared__ __align__(16) _Float16 lds_p[8 * 16 * 64];  // per-wave P tile

    const int tid  = threadIdx.x;
    const int lane = tid & 31;
    const int wv   = tid >> 5;
    const int h    = blockIdx.y;
    const int q0   = blockIdx.x * 128 + wv * 16;
    const int row_off = (lane & 16) ? 8 : 0;

    const _Float16* Qh = Q + (size_t)h * 4096 * 64;
    const _Float16* Kh = K + (size_t)h * 4096 * 64;
    const _Float16* Vh = V + (size_t)h * 4096 * 64;

    v16h qa0 = load_frag_a(Qh + (size_t)q0 * 64, 64, 0, 0, lane);
    v16h qa1 = load_frag_a(Qh + (size_t)q0 * 64, 64, 0, 32, lane);

    float m[8], l[8];
#pragma unroll
    for (int i = 0; i < 8; ++i) { m[i] = -3.0e38f; l[i] = 0.0f; }
    v8f o[4] = {};

    const float scale = 0.125f;        // 1/sqrt(64)
    const int nj = blockIdx.x * 2 + 2;

    for (int j = 0; j < nj; ++j) {
        const int kb = j * 64;
#if HAVE_TDM
        // K tile DMA: 64x64 f16, contiguous rows -> lds_k, overlapped with
        // the manual V^T staging below.
        if (wv == 0)
            tdm_load_2d_f16((unsigned int)(size_t)(void*)lds_k,
                            Kh + (size_t)kb * 64,
                            64, 4096, 64, 64, 64);
#else
        {   // stage K row-major (b128 copies)
            int r = tid >> 2, c = (tid & 3) * 16;
            const _Float16* gk = Kh + (size_t)(kb + r) * 64 + c;
            __builtin_prefetch(gk + 64 * 64, 0, 3);
            *(v8h*)(lds_k + r * 64 + c)     = *(const v8h*)gk;
            *(v8h*)(lds_k + r * 64 + c + 8) = *(const v8h*)(gk + 8);
        }
#endif
        {   // stage V transposed: key pairs packed -> ds_store_b32
            int r2 = (tid & 31) * 2;     // 0..62
            int c0 = (tid >> 5) * 8;     // 0..56
            const _Float16* g0 = Vh + (size_t)(kb + r2) * 64 + c0;
            v8h x0 = *(const v8h*)g0;
            v8h x1 = *(const v8h*)(g0 + 64);
            __builtin_prefetch(g0 + 64 * 64, 0, 3);
#pragma unroll
            for (int i = 0; i < 8; ++i) {
                v2h p;
                p[0] = x0[i];
                p[1] = x1[i];
                *(v2h*)(lds_vt + (c0 + i) * 64 + r2) = p;
            }
        }
#if HAVE_TDM
        if (wv == 0) __builtin_amdgcn_s_wait_tensorcnt(0);
#endif
        __syncthreads();

        // S = Q Kt
        v8f s[4];
#pragma unroll
        for (int nt = 0; nt < 4; ++nt) {
            v8f sa = {};
            v16h b0 = load_frag_b(lds_k, 64, nt * 16, 0, lane);
            sa = WMMA_F32_16x16x32_F16(qa0, b0, sa);
            v16h b1 = load_frag_b(lds_k, 64, nt * 16, 32, lane);
            sa = WMMA_F32_16x16x32_F16(qa1, b1, sa);
            s[nt] = sa;
        }

        // scale + causal mask
#pragma unroll
        for (int nt = 0; nt < 4; ++nt) {
            int key = kb + nt * 16 + (lane & 15);
#pragma unroll
            for (int i = 0; i < 8; ++i) {
                int qr = q0 + i + row_off;
                float sv = s[nt][i] * scale;
                s[nt][i] = (key <= qr) ? sv : -3.0e38f;
            }
        }

        // online softmax
        float mt[8];
#pragma unroll
        for (int i = 0; i < 8; ++i) {
            float v0 = fmaxf(fmaxf(s[0][i], s[1][i]), fmaxf(s[2][i], s[3][i]));
#pragma unroll
            for (int msk = 1; msk < 16; msk <<= 1)
                v0 = fmaxf(v0, __shfl_xor(v0, msk, 32));
            mt[i] = v0;
        }
#pragma unroll
        for (int i = 0; i < 8; ++i) {
            float mn    = fmaxf(m[i], mt[i]);
            float alpha = __expf(m[i] - mn);
            m[i] = mn;
            l[i] *= alpha;
#pragma unroll
            for (int nt = 0; nt < 4; ++nt) o[nt][i] *= alpha;
        }
#pragma unroll
        for (int i = 0; i < 8; ++i) {
            float rs = 0.0f;
#pragma unroll
            for (int nt = 0; nt < 4; ++nt) {
                float p = __expf(s[nt][i] - m[i]);
                s[nt][i] = p;
                rs += p;
            }
#pragma unroll
            for (int msk = 1; msk < 16; msk <<= 1)
                rs += __shfl_xor(rs, msk, 32);
            l[i] += rs;
        }

        // P: C layout -> A-fragment layout through per-wave LDS
        _Float16* pw = lds_p + wv * 16 * 64;
#pragma unroll
        for (int nt = 0; nt < 4; ++nt)
#pragma unroll
            for (int i = 0; i < 8; ++i)
                pw[(i + row_off) * 64 + nt * 16 + (lane & 15)] = (_Float16)s[nt][i];
        __syncthreads();

        // O += P V
#pragma unroll
        for (int ks = 0; ks < 2; ++ks) {
            v16h pa = load_frag_a(pw, 64, 0, ks * 32, lane);
#pragma unroll
            for (int nt = 0; nt < 4; ++nt) {
                v16h vb = load_frag_b(lds_vt, 64, nt * 16, ks * 32, lane);
                o[nt] = WMMA_F32_16x16x32_F16(pa, vb, o[nt]);
            }
        }
        __syncthreads();
    }

    // normalize, store y f16 [t][h*64+d]
#pragma unroll
    for (int i = 0; i < 8; ++i) {
        float invl = 1.0f / l[i];
        size_t t = (size_t)(q0 + i + row_off);
#pragma unroll
        for (int nt = 0; nt < 4; ++nt)
            Y[t * 1024 + h * 64 + nt * 16 + (lane & 15)] = (_Float16)(o[nt][i] * invl);
    }
}

// ---------------------------------------------------------------------------
// Kernel 3: out = y @ w_proj + b_proj (f32 output). 128x128 tiles.
// A tile (f16, strided) staged via TDM; B needs f32->f16 so staged manually.
__global__ __launch_bounds__(256) void proj_kernel(
        const _Float16* __restrict__ Y, const float* __restrict__ w,
        const float* __restrict__ b, float* __restrict__ out) {
    __shared__ __align__(16) _Float16 lds_a[128 * 32];
    __shared__ __align__(16) _Float16 lds_bt[128 * 32];

    const int tid  = threadIdx.x;
    const int lane = tid & 31;
    const int wv   = tid >> 5;
    const int m0   = blockIdx.y * 128;
    const int n0   = blockIdx.x * 128;
    const int mt   = (wv & 3) * 32;
    const int nb   = (wv >> 2) * 64;
    const int ncol = lane & 15;

    v8f acc[2][4] = {};

    for (int kk = 0; kk < 1024; kk += 32) {
#if HAVE_TDM
        // A tile DMA: 128 rows x 32 halves out of the 4096x1024 tensor.
        if (wv == 0)
            tdm_load_2d_f16((unsigned int)(size_t)(void*)lds_a,
                            Y + (size_t)m0 * 1024 + kk,
                            1024, 4096, 32, 128, 1024);
#else
        {   // stage A (already f16): straight b128 copies
            int r = tid >> 1, c = (tid & 1) * 16;
            const _Float16* gp = Y + (size_t)(m0 + r) * 1024 + kk + c;
            __builtin_prefetch(gp + 32, 0, 3);
            *(v8h*)(lds_a + r * 32 + c)     = *(const v8h*)gp;
            *(v8h*)(lds_a + r * 32 + c + 8) = *(const v8h*)(gp + 8);
        }
#endif
        {   // stage B transposed: pair of k-rows packed -> ds_store_b32
            int r  = (tid >> 4) * 2;
            int c0 = (tid & 15) * 8;
            const float* g0 = w + (size_t)(kk + r) * 1024 + n0 + c0;
            const float* g1 = g0 + 1024;
            __builtin_prefetch(g0 + 32 * 1024, 0, 3);
#pragma unroll
            for (int i = 0; i < 8; ++i) {
                v2h p;
                p[0] = (_Float16)g0[i];
                p[1] = (_Float16)g1[i];
                *(v2h*)(lds_bt + (c0 + i) * 32 + r) = p;
            }
        }
#if HAVE_TDM
        if (wv == 0) __builtin_amdgcn_s_wait_tensorcnt(0);
#endif
        __syncthreads();

        v16h a0 = load_frag_a(lds_a, 32, mt, 0, lane);
        v16h a1 = load_frag_a(lds_a, 32, mt + 16, 0, lane);
#pragma unroll
        for (int nt = 0; nt < 4; ++nt) {
            v16h bf = load_frag_b(lds_bt, 32, nb + nt * 16, 0, lane);
            acc[0][nt] = WMMA_F32_16x16x32_F16(a0, bf, acc[0][nt]);
            acc[1][nt] = WMMA_F32_16x16x32_F16(a1, bf, acc[1][nt]);
        }
        __syncthreads();
    }

    const int row_off = (lane & 16) ? 8 : 0;
#pragma unroll
    for (int rt = 0; rt < 2; ++rt) {
        const int mrow = m0 + mt + rt * 16 + row_off;
#pragma unroll
        for (int nt = 0; nt < 4; ++nt) {
            int col  = n0 + nb + nt * 16 + ncol;
            float bc = b[col];
#pragma unroll
            for (int i = 0; i < 8; ++i)
                out[(size_t)(mrow + i) * 1024 + col] = acc[rt][nt][i] + bc;
        }
    }
}

// ---------------------------------------------------------------------------
extern "C" void kernel_launch(void* const* d_in, const int* in_sizes, int n_in,
                              void* d_out, int out_size, void* d_ws, size_t ws_size,
                              hipStream_t stream) {
    (void)in_sizes; (void)n_in; (void)out_size; (void)ws_size;
    const float* x      = (const float*)d_in[0];
    const float* w_attn = (const float*)d_in[1];
    const float* b_attn = (const float*)d_in[2];
    const float* w_proj = (const float*)d_in[3];
    const float* b_proj = (const float*)d_in[4];
    float* out = (float*)d_out;

    const size_t HTD = (size_t)16 * 4096 * 64;  // per-tensor f16 elements
    _Float16* Qw = (_Float16*)d_ws;
    _Float16* Kw = Qw + HTD;
    _Float16* Vw = Kw + HTD;
    _Float16* Yw = Vw + HTD;

    // QKV GEMM + RoPE: M=4096 (128/blk), N=3072 (128/blk)
    qkv_rope_kernel<<<dim3(24, 32), 256, 0, stream>>>(x, w_attn, b_attn, Qw, Kw, Vw);
    // Flash attention: 32 query blocks x 16 heads
    attn_kernel<<<dim3(32, 16), 256, 0, stream>>>(Qw, Kw, Vw, Yw);
    // Projection: M=4096 (128/blk), N=1024 (128/blk)
    proj_kernel<<<dim3(8, 32), 256, 0, stream>>>(Yw, w_proj, b_proj, out);
}
